// CausalSelfAttention_61349312856509
// MI455X (gfx1250) — matrix operand order repaired and checked by reading, not compiled, not run on hardware
//
#include <hip/hip_runtime.h>
#include <math.h>

// ---------------- problem constants ----------------
#define BB 4
#define TT 2048
#define CC 1024
#define HH 16
#define DD 64
#define MM (BB*TT)          // 8192 rows
static constexpr float ATT_SCALE = 0.125f;   // 1/sqrt(64)

typedef __bf16 v16bf __attribute__((ext_vector_type(16)));
typedef __bf16 v8bf  __attribute__((ext_vector_type(8)));
typedef float  v8f   __attribute__((ext_vector_type(8)));

__device__ __forceinline__ __bf16 f2bf(float f) {
    return (__bf16)f;                         // native v_cvt bf16 path
}

__device__ __forceinline__ v16bf cat8(v8bf lo, v8bf hi) {
    return __builtin_shufflevector(lo, hi, 0,1,2,3,4,5,6,7,8,9,10,11,12,13,14,15);
}

// LDS byte-offset of a generic pointer to a __shared__ object
// (flat LDS aperture keeps the offset in addr[31:0]).
__device__ __forceinline__ unsigned lds_off(const void* p) {
    return (unsigned)(unsigned long long)p;
}

// Async copy 16B global -> LDS, tracked on ASYNCcnt.
__device__ __forceinline__ void async_copy_b128(unsigned lds_byte_off,
                                                const void* gaddr) {
    asm volatile("global_load_async_to_lds_b128 %0, %1, off"
                 :: "v"(lds_byte_off), "v"((unsigned long long)gaddr)
                 : "memory");
}

__device__ __forceinline__ void wait_async0() {
    asm volatile("s_wait_asynccnt 0x0" ::: "memory");
}

// =====================================================================
// GEMM: Y[M,N] = X[M,K] @ W[K,N] + bias, K=N=1024, M=8192.
// 128x128 tile / workgroup, 8 wave32 waves (4x2), BK=32 per WMMA step.
// IN_BF16:   X is bf16 (attention output) -> async copy to LDS;
//            else f32 -> VGPR convert to bf16.
// OUT_HEADS: write bf16 to [B,H,T,D]; else write f32 to [M,N].
// =====================================================================
template<bool IN_BF16, bool OUT_HEADS>
__global__ __launch_bounds__(256)
void gemm128(const float* __restrict__ Xf, const __bf16* __restrict__ Xb,
             const float* __restrict__ W,  const float* __restrict__ bias,
             __bf16* __restrict__ Yb, float* __restrict__ Yf)
{
    constexpr int BM = 128, BN = 128, BK = 32, LDA = BK + 8;   // 40*2B rows: 16B aligned
    __shared__ __bf16 As[BM * LDA];   // [m][k]
    __shared__ __bf16 Bs[BN * LDA];   // [n][k]  (W tile transposed)

    const int tid  = threadIdx.x;
    const int lane = tid & 31, wave = tid >> 5;
    const int h    = lane >> 4, lm = lane & 15;
    const int wm   = wave >> 1, wn = wave & 1;
    const int tileN = blockIdx.x * BN;
    const int tileM = blockIdx.y * BM;

    v8f acc[2][4];
#pragma unroll
    for (int a = 0; a < 2; ++a)
#pragma unroll
        for (int b = 0; b < 4; ++b)
#pragma unroll
            for (int e = 0; e < 8; ++e) acc[a][b][e] = 0.f;

    for (int k0 = 0; k0 < CC; k0 += BK) {
        __syncthreads();
        // ---- stage A tile (128 x 32) ----
        if constexpr (!IN_BF16) {
#pragma unroll
            for (int it = 0; it < 4; ++it) {
                int lin = it * 256 + tid;            // 0..1023
                int r = lin >> 3;                    // 0..127
                int c = (lin & 7) * 4;
                const float4 v4 = *(const float4*)(Xf + (size_t)(tileM + r) * CC + k0 + c);
                __bf16* dst = &As[r * LDA + c];
                dst[0] = f2bf(v4.x); dst[1] = f2bf(v4.y);
                dst[2] = f2bf(v4.z); dst[3] = f2bf(v4.w);
            }
            if (k0 + BK < CC)
                __builtin_prefetch(Xf + (size_t)(tileM + (tid >> 1)) * CC + k0 + BK, 0, 1);
        } else {
            // async bf16 copy global -> LDS (no VGPR round-trip)
#pragma unroll
            for (int it = 0; it < 2; ++it) {
                int lin = it * 256 + tid;            // 0..511
                int r = lin >> 2;                    // 0..127
                int c = (lin & 3) * 8;
                async_copy_b128(lds_off(&As[r * LDA + c]),
                                Xb + (size_t)(tileM + r) * CC + k0 + c);
            }
            if (k0 + BK < CC)
                __builtin_prefetch(Xb + (size_t)(tileM + (tid >> 1)) * CC + k0 + BK, 0, 1);
        }
        // ---- stage W tile (32 x 128), transposed into Bs[n][k] ----
#pragma unroll
        for (int it = 0; it < 4; ++it) {
            int lin = it * 256 + tid;                // 0..1023
            int kk = lin >> 5;                       // 0..31
            int nn = (lin & 31) * 4;
            const float4 v4 = *(const float4*)(W + (size_t)(k0 + kk) * CC + tileN + nn);
            Bs[(nn + 0) * LDA + kk] = f2bf(v4.x);
            Bs[(nn + 1) * LDA + kk] = f2bf(v4.y);
            Bs[(nn + 2) * LDA + kk] = f2bf(v4.z);
            Bs[(nn + 3) * LDA + kk] = f2bf(v4.w);
        }
        if constexpr (IN_BF16) wait_async0();
        __syncthreads();

        // ---- fragments + WMMA ----
        v16bf afrag[2], bfrag[4];
#pragma unroll
        for (int fm = 0; fm < 2; ++fm) {
            const __bf16* row = &As[(wm * 32 + fm * 16 + lm) * LDA];
            v8bf c0 = *(const v8bf*)(row + 8 * h);        // K = 8h..8h+7
            v8bf c1 = *(const v8bf*)(row + 16 + 8 * h);   // K = 16+8h..
            afrag[fm] = cat8(c0, c1);
        }
#pragma unroll
        for (int fn = 0; fn < 4; ++fn) {
            const __bf16* row = &Bs[(wn * 64 + fn * 16 + lm) * LDA];
            v8bf c0 = *(const v8bf*)(row + 8 * h);        // K = 8h..8h+7
            v8bf c1 = *(const v8bf*)(row + 16 + 8 * h);   // K = 16+8h..
            bfrag[fn] = cat8(c0, c1);
        }
#pragma unroll
        for (int fm = 0; fm < 2; ++fm)
#pragma unroll
            for (int fn = 0; fn < 4; ++fn)
                acc[fm][fn] = __builtin_amdgcn_wmma_f32_16x16x32_bf16(
                    false, afrag[fm], false, bfrag[fn],
                    (short)0, acc[fm][fn], false, false);
    }

    // ---- epilogue: bias + store ----
#pragma unroll
    for (int fm = 0; fm < 2; ++fm)
#pragma unroll
        for (int fn = 0; fn < 4; ++fn) {
            const int colg = tileN + wn * 64 + fn * 16 + lm;
            const float bv = bias[colg];
#pragma unroll
            for (int i = 0; i < 8; ++i) {
                const int rowg = tileM + wm * 32 + fm * 16 + i + 8 * h;
                const float v = acc[fm][fn][i] + bv;
                if constexpr (OUT_HEADS) {
                    const int b  = rowg >> 11;          // /T
                    const int t  = rowg & (TT - 1);
                    const int hd = colg >> 6;           // /D
                    const int d  = colg & (DD - 1);
                    Yb[(size_t)((b * HH + hd) * TT + t) * DD + d] = f2bf(v);
                } else {
                    Yf[(size_t)rowg * CC + colg] = v;
                }
            }
        }
}

// =====================================================================
// Flash attention: per (b,h) head, Br=128 query rows / WG (8 waves x 16
// rows), Bc=64 key tile, online softmax f32, bf16 WMMA for QK^T and PV.
// Q/K tiles staged with GLOBAL_LOAD_ASYNC_TO_LDS_B128 (ASYNCcnt);
// V staged through VGPRs (needs [d][tk] transpose).
// Causal: tile loop bounded at the diagonal + per-element mask on edge.
// =====================================================================
__global__ __launch_bounds__(256)
void flash_attn(const __bf16* __restrict__ Qh, const __bf16* __restrict__ Kh,
                const __bf16* __restrict__ Vh, __bf16* __restrict__ Ob)
{
    constexpr int BR = 128, BC = 64;
    constexpr int LDQ = DD + 8, LDK = DD + 8, LDV = BC + 8, LDP = BC + 8;
    __shared__ __bf16 Qs[BR * LDQ];
    __shared__ __bf16 Ks[BC * LDK];   // [tk][d]
    __shared__ __bf16 Vs[DD * LDV];   // [d][tk]  (transposed)
    __shared__ __bf16 Ps[BR * LDP];   // [r][tk]

    const int tid  = threadIdx.x, lane = tid & 31, wave = tid >> 5;
    const int h    = lane >> 4, lm = lane & 15;
    const int q0   = blockIdx.x * BR;
    const int bh   = blockIdx.y;                       // b*H + head
    const size_t base = (size_t)bh * TT * DD;
    const int wbase = wave * 16;

    // ---- async load Q tile once ----
#pragma unroll
    for (int it = 0; it < 4; ++it) {
        int lin = it * 256 + tid;      // 0..1023
        int r = lin >> 3;              // 0..127
        int c = (lin & 7) * 8;
        async_copy_b128(lds_off(&Qs[r * LDQ + c]),
                        Qh + base + (size_t)(q0 + r) * DD + c);
    }

    float mrow[8], lrow[8];
    v8f o[4];
#pragma unroll
    for (int i = 0; i < 8; ++i) { mrow[i] = -1e30f; lrow[i] = 0.f; }
#pragma unroll
    for (int fd = 0; fd < 4; ++fd)
#pragma unroll
        for (int i = 0; i < 8; ++i) o[fd][i] = 0.f;

    const int ntiles = (q0 + BR) / BC;                 // causal bound

    for (int tile = 0; tile < ntiles; ++tile) {
        const int kv0 = tile * BC;
        __syncthreads();
        // ---- stage K async (straight) and V through VGPRs (transposed) ----
#pragma unroll
        for (int it = 0; it < 2; ++it) {
            int lin = it * 256 + tid;  // 0..511
            int tk = lin >> 3;         // 0..63
            int d0 = (lin & 7) * 8;
            async_copy_b128(lds_off(&Ks[tk * LDK + d0]),
                            Kh + base + (size_t)(kv0 + tk) * DD + d0);
            v8bf vv = *(const v8bf*)(Vh + base + (size_t)(kv0 + tk) * DD + d0);
#pragma unroll
            for (int j = 0; j < 8; ++j) Vs[(d0 + j) * LDV + tk] = vv[j];
        }
        if (tile + 1 < ntiles) {
            __builtin_prefetch(Kh + base + (size_t)(kv0 + BC + (tid >> 2)) * DD, 0, 1);
            __builtin_prefetch(Vh + base + (size_t)(kv0 + BC + (tid >> 2)) * DD, 0, 1);
        }
        wait_async0();                 // Q (first iter) + K copies complete
        __syncthreads();

        // ---- S = Q K^T  (16 x 64 per wave) ----
        v8f s[4];
#pragma unroll
        for (int fn = 0; fn < 4; ++fn)
#pragma unroll
            for (int i = 0; i < 8; ++i) s[fn][i] = 0.f;
#pragma unroll
        for (int kd = 0; kd < 2; ++kd) {
            const __bf16* qrow = &Qs[(wbase + lm) * LDQ + kd * 32];
            v16bf afrag = cat8(*(const v8bf*)(qrow + 8 * h),
                               *(const v8bf*)(qrow + 16 + 8 * h));
#pragma unroll
            for (int fn = 0; fn < 4; ++fn) {
                const __bf16* krow = &Ks[(fn * 16 + lm) * LDK + kd * 32 + 8 * h];
                v16bf bfrag = cat8(*(const v8bf*)(krow),
                                   *(const v8bf*)(krow + 16));
                s[fn] = __builtin_amdgcn_wmma_f32_16x16x32_bf16(
                    false, afrag, false, bfrag, (short)0, s[fn], false, false);
            }
        }

        // ---- scale + causal mask (wave-uniform branch) ----
        const bool needMask = (kv0 + BC) > (q0 + wbase);
#pragma unroll
        for (int fn = 0; fn < 4; ++fn)
#pragma unroll
            for (int i = 0; i < 8; ++i) {
                float sv = s[fn][i] * ATT_SCALE;
                if (needMask) {
                    const int rg = q0 + wbase + i + 8 * h;
                    const int cg = kv0 + fn * 16 + lm;
                    if (cg > rg) sv = -1e30f;
                }
                s[fn][i] = sv;
            }

        // ---- online softmax (rows live in 16-lane groups) ----
#pragma unroll
        for (int i = 0; i < 8; ++i) {
            float mloc = fmaxf(fmaxf(s[0][i], s[1][i]), fmaxf(s[2][i], s[3][i]));
#pragma unroll
            for (int off = 1; off < 16; off <<= 1)
                mloc = fmaxf(mloc, __shfl_xor(mloc, off, 32));
            const float mnew = fmaxf(mrow[i], mloc);
            const float corr = __expf(mrow[i] - mnew);
            float psum = 0.f;
#pragma unroll
            for (int fn = 0; fn < 4; ++fn) {
                const float p = __expf(s[fn][i] - mnew);
                s[fn][i] = p;
                psum += p;
            }
#pragma unroll
            for (int off = 1; off < 16; off <<= 1)
                psum += __shfl_xor(psum, off, 32);
            lrow[i] = lrow[i] * corr + psum;
            mrow[i] = mnew;
#pragma unroll
            for (int fd = 0; fd < 4; ++fd) o[fd][i] *= corr;
        }

        // ---- P -> LDS (wave-local rows; re-layout C-frag -> A-frag) ----
#pragma unroll
        for (int fn = 0; fn < 4; ++fn)
#pragma unroll
            for (int i = 0; i < 8; ++i)
                Ps[(wbase + i + 8 * h) * LDP + fn * 16 + lm] = f2bf(s[fn][i]);

        // ---- O += P V ----
#pragma unroll
        for (int kt = 0; kt < 2; ++kt) {
            const __bf16* prow = &Ps[(wbase + lm) * LDP + kt * 32];
            v16bf pfrag = cat8(*(const v8bf*)(prow + 8 * h),
                               *(const v8bf*)(prow + 16 + 8 * h));
#pragma unroll
            for (int fd = 0; fd < 4; ++fd) {
                const __bf16* vrow = &Vs[(fd * 16 + lm) * LDV + kt * 32 + 8 * h];
                v16bf vfrag = cat8(*(const v8bf*)(vrow),
                                   *(const v8bf*)(vrow + 16));
                o[fd] = __builtin_amdgcn_wmma_f32_16x16x32_bf16(
                    false, pfrag, false, vfrag, (short)0, o[fd], false, false);
            }
        }
    }

    // ---- epilogue: O / l, scatter to [M, C] bf16 ----
    const int b = bh / HH, hd = bh % HH;
#pragma unroll
    for (int fd = 0; fd < 4; ++fd)
#pragma unroll
        for (int i = 0; i < 8; ++i) {
            const int rg  = q0 + wbase + i + 8 * h;
            const int col = hd * DD + fd * 16 + lm;
            const float val = o[fd][i] / lrow[i];
            Ob[(size_t)(b * TT + rg) * CC + col] = f2bf(val);
        }
}

// =====================================================================
// Host launch
// =====================================================================
extern "C" void kernel_launch(void* const* d_in, const int* in_sizes, int n_in,
                              void* d_out, int out_size, void* d_ws, size_t ws_size,
                              hipStream_t stream)
{
    const float* k  = (const float*)d_in[0];
    const float* v  = (const float*)d_in[1];
    const float* q  = (const float*)d_in[2];
    const float* Wq = (const float*)d_in[3];
    const float* bq = (const float*)d_in[4];
    const float* Wk = (const float*)d_in[5];
    const float* bk = (const float*)d_in[6];
    const float* Wv = (const float*)d_in[7];
    const float* bv = (const float*)d_in[8];
    const float* Wo = (const float*)d_in[9];
    const float* bo = (const float*)d_in[10];
    float* out = (float*)d_out;

    // workspace layout: 4 x 16 MB of bf16 (qh, kh, vh, attn-out)
    char* wsb = (char*)d_ws;
    const size_t seg = (size_t)MM * CC * sizeof(__bf16);   // 16 MB
    __bf16* qh = (__bf16*)(wsb + 0 * seg);   // [B,H,T,D]
    __bf16* kh = (__bf16*)(wsb + 1 * seg);   // [B,H,T,D]
    __bf16* vh = (__bf16*)(wsb + 2 * seg);   // [B,H,T,D]
    __bf16* ob = (__bf16*)(wsb + 3 * seg);   // [M, C]
    (void)in_sizes; (void)n_in; (void)out_size; (void)ws_size;

    dim3 gGemm(CC / 128, MM / 128);   // (8, 64)
    dim3 blk(256);

    gemm128<false, true><<<gGemm, blk, 0, stream>>>(q, nullptr, Wq, bq, qh, nullptr);
    gemm128<false, true><<<gGemm, blk, 0, stream>>>(k, nullptr, Wk, bk, kh, nullptr);
    gemm128<false, true><<<gGemm, blk, 0, stream>>>(v, nullptr, Wv, bv, vh, nullptr);

    dim3 gAttn(TT / 128, BB * HH);    // (16, 64)
    flash_attn<<<gAttn, blk, 0, stream>>>(qh, kh, vh, ob);

    gemm128<true, false><<<gGemm, blk, 0, stream>>>(nullptr, ob, Wo, bo, nullptr, out);
}
